// Tacotron_35923106463918
// MI455X (gfx1250) — matrix operand order repaired and checked by reading, not compiled, not run on hardware
//
#include <hip/hip_runtime.h>

#define B_ 32
#define L_ 256
#define T_ 400
#define ENC_ 512
#define DEC_ 1024
#define PRE_ 256
#define ATT_ 128
#define NMEL_ 80
#define GRID_ 64
#define BLOCK_ 256

typedef __attribute__((ext_vector_type(16))) __bf16 v16bf;
typedef __attribute__((ext_vector_type(8)))  float  v8f;
typedef __attribute__((ext_vector_type(4)))  unsigned u32x4;

union Frag {
  v16bf v;
  unsigned short u[16];
  u32x4 q[2];
};

__device__ inline unsigned short f2bf(float f){
  union { float f; unsigned u; } c; c.f = f;
  unsigned u = c.u;
  u += 0x7fffu + ((u >> 16) & 1u);           // round-to-nearest-even
  return (unsigned short)(u >> 16);
}
__device__ inline float sigmf(float x){ return 1.0f / (1.0f + __expf(-x)); }
__device__ inline unsigned hash_u32(unsigned x){
  x ^= x >> 16; x *= 0x7feb352du; x ^= x >> 15; x *= 0x846ca68bu; x ^= x >> 16;
  return x;
}

// A fragment (16x32 bf16) from row-major bf16 activations, per ISA 7.12.2:
// lanes 0-15: M=lane, K = kb+{0..7, 16..23}; lanes 16-31: M=lane-16, K = kb+{8..15, 24..31}
__device__ inline void load_a_bf16(Frag& A, const unsigned short* act, int Kdim, int row0, int kb){
  int lane = threadIdx.x & 31;
  int base = (lane >> 4) << 3;     // 0 or 8
  const unsigned short* p = act + (long)(row0 + (lane & 15)) * Kdim + kb + base;
  A.q[0] = *(const u32x4*)p;        // K = kb+base .. +7
  A.q[1] = *(const u32x4*)(p + 16); // K = kb+16+base .. +7
}
// B fragment (32x16 bf16) from pre-packed layout: 32 lanes x 16 contiguous bf16
__device__ inline void load_b_frag(Frag& Bf, const unsigned short* pw){
  int lane = threadIdx.x & 31;
  const unsigned short* p = pw + (lane << 4);
  Bf.q[0] = *(const u32x4*)p;
  Bf.q[1] = *(const u32x4*)(p + 8);
}
__device__ inline v8f wmma_bf16(const Frag& A, const Frag& Bf, v8f C){
  return __builtin_amdgcn_wmma_f32_16x16x32_bf16(false, A.v, false, Bf.v, (short)0, C, false, false);
}

// ---------------- weight pre-pack: fp32 [K,N] row-major -> bf16 WMMA-B fragments --------
__global__ void pack_w_bf16(const float* __restrict__ src, unsigned short* __restrict__ dst,
                            int Kpad, int Kreal, int N){
  int ktiles = Kpad >> 5, ntiles = N >> 4;
  long total = (long)ktiles * ntiles * 512;
  long idx = (long)blockIdx.x * blockDim.x + threadIdx.x;
  if (idx >= total) return;
  int e    = (int)(idx & 15);
  int lane = (int)((idx >> 4) & 31);
  long tile = idx >> 9;
  int nt = (int)(tile % ntiles), kt = (int)(tile / ntiles);
  int kbase = (lane >= 16) ? 8 : 0;
  int k = kt * 32 + kbase + (e & 7) + ((e >> 3) << 4);
  int n = nt * 16 + (lane & 15);
  float v = (k < Kreal) ? src[(long)k * N + n] : 0.0f;
  dst[idx] = f2bf(v);
}

__global__ void zero_kernel(unsigned* __restrict__ p, long nwords){
  long i = (long)blockIdx.x * blockDim.x + threadIdx.x;
  if (i < nwords) p[i] = 0u;
}

// ---------------- prenet: rows r = t*32+b (t=0..399), x = zero-frame or target ----------
__global__ void __launch_bounds__(256) prenet_kernel(
    const float* __restrict__ target, const float* __restrict__ bp1, const float* __restrict__ bp2,
    const unsigned short* __restrict__ pWp1, const unsigned short* __restrict__ pWp2,
    unsigned short* __restrict__ tpre)
{
  __shared__ __align__(16) float sx[16 * 96];
  __shared__ __align__(16) unsigned short sh1[16 * 256];
  int tile = blockIdx.x;                 // 800 tiles of 16 rows
  int tid = threadIdx.x;
  for (int j = tid; j < 16 * 96; j += 256){
    int lr = j / 96, kk = j % 96;
    int r = tile * 16 + lr, t = r >> 5, b = r & 31;
    float v = 0.0f;
    if (kk < NMEL_ && t > 0) v = target[((long)b * NMEL_ + kk) * T_ + (t - 1)];
    sx[j] = v;
  }
  __syncthreads();
  int wv = tid >> 5, lane = tid & 31;
  int base = (lane >> 4) << 3, arow = lane & 15;
  // layer 1: K=96 (pad of 80), N=256
  for (int nn = 0; nn < 2; ++nn){
    int nt = wv * 2 + nn;
    v8f acc = {};
    for (int kt = 0; kt < 3; ++kt){
      Frag A, Bf;
      int kb = kt * 32;
      #pragma unroll
      for (int e = 0; e < 8; ++e){
        A.u[e]     = f2bf(sx[arow * 96 + kb + base + e]);
        A.u[e + 8] = f2bf(sx[arow * 96 + kb + 16 + base + e]);
      }
      load_b_frag(Bf, pWp1 + ((long)(kt * 16 + nt) << 9));
      acc = wmma_bf16(A, Bf, acc);
    }
    int rr0 = (lane >> 4) * 8, col = lane & 15;
    #pragma unroll
    for (int i = 0; i < 8; ++i){
      int row = rr0 + i, n = nt * 16 + col, r = tile * 16 + row;
      float h = acc[i] + bp1[n];
      h = h > 0.0f ? h : 0.0f;
      unsigned key = hash_u32(((unsigned)r * 256u + (unsigned)n) ^ 0xA5A5A5A5u);
      h = (key & 1u) ? h * 2.0f : 0.0f;  // dropout p=0.5, scale 2
      sh1[row * 256 + n] = f2bf(h);
    }
  }
  __syncthreads();
  // layer 2: K=256, N=256 (dual accumulators break the WMMA C->D chain)
  for (int nn = 0; nn < 2; ++nn){
    int nt = wv * 2 + nn;
    v8f acc0 = {}, acc1 = {};
    for (int kt = 0; kt < 8; kt += 2){
      Frag A0, B0, A1, B1;
      load_a_bf16(A0, sh1, 256, 0, kt * 32);
      load_b_frag(B0, pWp2 + ((long)(kt * 16 + nt) << 9));
      load_a_bf16(A1, sh1, 256, 0, (kt + 1) * 32);
      load_b_frag(B1, pWp2 + ((long)((kt + 1) * 16 + nt) << 9));
      acc0 = wmma_bf16(A0, B0, acc0);
      acc1 = wmma_bf16(A1, B1, acc1);
    }
    v8f acc = acc0 + acc1;
    int rr0 = (lane >> 4) * 8, col = lane & 15;
    #pragma unroll
    for (int i = 0; i < 8; ++i){
      int row = rr0 + i, n = nt * 16 + col, r = tile * 16 + row;
      float h = acc[i] + bp2[n];
      h = h > 0.0f ? h : 0.0f;
      unsigned key = hash_u32(((unsigned)r * 256u + (unsigned)n) ^ 0x5A5A5A5Au);
      h = (key & 1u) ? h * 2.0f : 0.0f;
      tpre[(long)r * 256 + n] = f2bf(h);
    }
  }
}

// ---------------- pmem = enc[8192,512] @ Wm[512,128], stored transposed [b][a][l] -------
__global__ void __launch_bounds__(256) pmem_kernel(
    const float* __restrict__ enc, const unsigned short* __restrict__ pWm,
    float* __restrict__ pmem_t)
{
  int tile = blockIdx.x;                 // 512 tiles of 16 rows
  int wv = threadIdx.x >> 5, lane = threadIdx.x & 31;
  int nt = wv;                           // 8 ntiles of 16 -> ATT=128
  int base = (lane >> 4) << 3;
  int arow = tile * 16 + (lane & 15);
  v8f acc0 = {}, acc1 = {};
  for (int kt = 0; kt < 16; kt += 2){
    Frag A0, B0, A1, B1;
    const float* p0 = enc + (long)arow * ENC_ + kt * 32 + base;
    const float* p1 = enc + (long)arow * ENC_ + (kt + 1) * 32 + base;
    #pragma unroll
    for (int e = 0; e < 8; ++e){
      A0.u[e] = f2bf(p0[e]); A0.u[e + 8] = f2bf(p0[16 + e]);
      A1.u[e] = f2bf(p1[e]); A1.u[e + 8] = f2bf(p1[16 + e]);
    }
    load_b_frag(B0, pWm + ((long)(kt * 8 + nt) << 9));
    load_b_frag(B1, pWm + ((long)((kt + 1) * 8 + nt) << 9));
    acc0 = wmma_bf16(A0, B0, acc0);
    acc1 = wmma_bf16(A1, B1, acc1);
  }
  v8f acc = acc0 + acc1;
  int rr0 = (lane >> 4) * 8, col = lane & 15;
  #pragma unroll
  for (int i = 0; i < 8; ++i){
    int r = tile * 16 + rr0 + i;         // r = b*256 + l
    int b = r >> 8, l = r & 255, a = nt * 16 + col;
    pmem_t[((long)b * ATT_ + a) * L_ + l] = acc[i];
  }
}

// ---------------- persistent decoder: 64 WGs x 256 threads, grid barriers ---------------
__device__ inline void grid_barrier(unsigned* bar){
  __threadfence();
  __syncthreads();
  if (threadIdx.x == 0){
    unsigned gen = atomicAdd(bar + 1, 0u);
    unsigned arrived = atomicAdd(bar, 1u) + 1u;
    if (arrived == gridDim.x){
      atomicExch(bar, 0u);
      __threadfence();
      atomicAdd(bar + 1, 1u);
    } else {
      while (atomicAdd(bar + 1, 0u) == gen) __builtin_amdgcn_s_sleep(2);
    }
  }
  __syncthreads();
  __threadfence();
}

// one dual-issue GEMM slice: acc0 += A(kt)·B(kt), acc1 += A(kt+1)·B(kt+1)
__device__ inline void gemm_pair(v8f& acc0, v8f& acc1,
                                 const unsigned short* act, int Kdim, int row0, int kt,
                                 const unsigned short* pw, int ntiles, int nt){
  Frag A0, B0, A1, B1;
  load_a_bf16(A0, act, Kdim, row0, kt * 32);
  load_b_frag(B0, pw + ((long)(kt * ntiles + nt) << 9));
  load_a_bf16(A1, act, Kdim, row0, (kt + 1) * 32);
  load_b_frag(B1, pw + ((long)((kt + 1) * ntiles + nt) << 9));
  acc0 = wmma_bf16(A0, B0, acc0);
  acc1 = wmma_bf16(A1, B1, acc1);
}

__global__ void __launch_bounds__(256) decoder_kernel(
    const float* __restrict__ enc, const int* __restrict__ lens,
    const float* __restrict__ b_att, const float* __restrict__ b_gen,
    const float* __restrict__ Wq, const float* __restrict__ loc_conv,
    const float* __restrict__ Wl, const float* __restrict__ v_att,
    const float* __restrict__ Wf, const float* __restrict__ bf_,
    const float* __restrict__ Ws, const float* __restrict__ bs_,
    const unsigned short* __restrict__ pWx_att, const unsigned short* __restrict__ pWh_att,
    const unsigned short* __restrict__ pWx_gen, const unsigned short* __restrict__ pWh_gen,
    const unsigned short* __restrict__ tpre, const float* __restrict__ pmem_t,
    float* gates, float* h_att, float* c_att, float* h_gen, float* c_gen,
    unsigned short* hatt_bf, unsigned short* hgen_bf,
    float* ctx_f, unsigned short* ctx_bf,
    float* prev_w, float* cum_w, unsigned* bar,
    float* out_spec, float* out_stop, float* out_align)
{
  __shared__ float sWl[32 * 128];
  __shared__ float sconv[32 * 62];
  __shared__ float sv[128];
  __shared__ float sq[128];
  __shared__ float sprev[286];
  __shared__ float scum[286];
  __shared__ float sred[256];

  const int wg = blockIdx.x, tid = threadIdx.x;
  const int wv = tid >> 5, lane = tid & 31;
  const int gw = wg * 8 + wv;                       // 0..511: one 16x16 gate tile per wave
  const int nt = gw & 255, mt = gw >> 8;
  const int rr0 = (lane >> 4) * 8, col = lane & 15;

  if (wg < B_){                                     // stage attention constants once
    for (int j = tid; j < 32 * 128; j += 256) sWl[j] = Wl[j];
    for (int j = tid; j < 32 * 62; j += 256) sconv[j] = loc_conv[j];
    if (tid < 128) sv[tid] = v_att[tid];
    __syncthreads();
  }

  #pragma unroll 1
  for (int t = 0; t < T_; ++t){
    // ---- P1: att-LSTM gates = tpre_t·Wx[0:256] + ctx·Wx[256:768] + h_att·Wh + b ----
    {
      v8f acc0 = {}, acc1 = {};
      const unsigned short* tp = tpre + (long)t * B_ * PRE_;
      for (int kt = 0; kt < 8; kt += 2)
        gemm_pair(acc0, acc1, tp, PRE_, mt * 16, kt, pWx_att, 256, nt);
      for (int kt = 0; kt < 16; kt += 2)
        gemm_pair(acc0, acc1, ctx_bf, ENC_, mt * 16, kt, pWx_att + ((long)(8 * 256) << 9), 256, nt);
      for (int kt = 0; kt < 32; kt += 2)
        gemm_pair(acc0, acc1, hatt_bf, DEC_, mt * 16, kt, pWh_att, 256, nt);
      v8f acc = acc0 + acc1;
      int n = nt * 16 + col;
      float bia = b_att[n];
      #pragma unroll
      for (int i = 0; i < 8; ++i)
        gates[(long)(mt * 16 + rr0 + i) * 4096 + n] = acc[i] + bia;
    }
    grid_barrier(bar);
    // ---- P2: att-LSTM pointwise -> h_att, c_att ----
    for (int idx = wg * 256 + tid; idx < B_ * DEC_; idx += GRID_ * BLOCK_){
      int b = idx >> 10, u = idx & 1023;
      const float* g = gates + (long)b * 4096;
      float c = sigmf(g[1024 + u]) * c_att[idx] + sigmf(g[u]) * tanhf(g[2048 + u]);
      float h = sigmf(g[3072 + u]) * tanhf(c);
      c_att[idx] = c; h_att[idx] = h; hatt_bf[idx] = f2bf(h);
    }
    grid_barrier(bar);
    // ---- P3: location-sensitive attention (one WG per batch row) ----
    if (wg < B_){
      int b = wg;
      for (int a = tid; a < ATT_; a += 256){        // q = h_att[b]·Wq
        float s = 0.0f;
        const float* hr = h_att + (long)b * DEC_;
        for (int k = 0; k < DEC_; ++k) s += hr[k] * Wq[(long)k * ATT_ + a];
        sq[a] = s;
      }
      for (int j = tid; j < 286; j += 256){         // halo-staged alignments
        int l = j - 15;
        bool ok = (l >= 0 && l < L_);
        sprev[j] = ok ? prev_w[b * L_ + l] : 0.0f;
        scum[j]  = ok ? cum_w[b * L_ + l] : 0.0f;
      }
      __syncthreads();
      int l = tid;
      float feat[32];
      #pragma unroll
      for (int f = 0; f < 32; ++f) feat[f] = 0.0f;
      for (int k = 0; k < 31; ++k){                 // SAME conv, kernel 31, 2 in-ch
        float p = sprev[l + k], c = scum[l + k];
        #pragma unroll
        for (int f = 0; f < 32; ++f)
          feat[f] += p * sconv[f * 62 + k] + c * sconv[f * 62 + 31 + k];
      }
      float e = 0.0f;
      const float* pm = pmem_t + (long)b * ATT_ * L_;
      for (int a = 0; a < ATT_; ++a){
        float lf = 0.0f;
        #pragma unroll
        for (int f = 0; f < 32; ++f) lf += feat[f] * sWl[f * 128 + a];
        e += sv[a] * tanhf(sq[a] + pm[(long)a * L_ + l] + lf);
      }
      if (l >= lens[b]) e = -1e9f;
      sred[tid] = e; __syncthreads();               // softmax over L
      for (int s2 = 128; s2 > 0; s2 >>= 1){
        if (tid < s2) sred[tid] = fmaxf(sred[tid], sred[tid + s2]);
        __syncthreads();
      }
      float mx = sred[0]; __syncthreads();
      float ex = __expf(e - mx);
      sred[tid] = ex; __syncthreads();
      for (int s2 = 128; s2 > 0; s2 >>= 1){
        if (tid < s2) sred[tid] += sred[tid + s2];
        __syncthreads();
      }
      float w = ex / sred[0];
      prev_w[b * L_ + l] = w;
      cum_w[b * L_ + l] += w;
      out_align[((long)b * T_ + t) * L_ + l] = w;
    }
    grid_barrier(bar);
    // ---- P4: context = w · enc ----
    {
      int idx = wg * 256 + tid;                     // exactly 32*512
      int b = idx >> 9, e = idx & 511;
      const float* wrow = prev_w + b * L_;
      const float* er = enc + (long)b * L_ * ENC_ + e;
      float s = 0.0f;
      for (int l = 0; l < L_; ++l) s += wrow[l] * er[(long)l * ENC_];
      ctx_f[idx] = s; ctx_bf[idx] = f2bf(s);
    }
    grid_barrier(bar);
    // ---- P5: gen-LSTM gates = h_att·Wx[0:1024] + ctx·Wx[1024:1536] + h_gen·Wh + b ----
    {
      v8f acc0 = {}, acc1 = {};
      for (int kt = 0; kt < 32; kt += 2)
        gemm_pair(acc0, acc1, hatt_bf, DEC_, mt * 16, kt, pWx_gen, 256, nt);
      for (int kt = 0; kt < 16; kt += 2)
        gemm_pair(acc0, acc1, ctx_bf, ENC_, mt * 16, kt, pWx_gen + ((long)(32 * 256) << 9), 256, nt);
      for (int kt = 0; kt < 32; kt += 2)
        gemm_pair(acc0, acc1, hgen_bf, DEC_, mt * 16, kt, pWh_gen, 256, nt);
      v8f acc = acc0 + acc1;
      int n = nt * 16 + col;
      float bia = b_gen[n];
      #pragma unroll
      for (int i = 0; i < 8; ++i)
        gates[(long)(mt * 16 + rr0 + i) * 4096 + n] = acc[i] + bia;
    }
    grid_barrier(bar);
    // ---- P6: gen-LSTM pointwise -> h_gen, c_gen ----
    for (int idx = wg * 256 + tid; idx < B_ * DEC_; idx += GRID_ * BLOCK_){
      int b = idx >> 10, u = idx & 1023;
      const float* g = gates + (long)b * 4096;
      float c = sigmf(g[1024 + u]) * c_gen[idx] + sigmf(g[u]) * tanhf(g[2048 + u]);
      float h = sigmf(g[3072 + u]) * tanhf(c);
      c_gen[idx] = c; h_gen[idx] = h; hgen_bf[idx] = f2bf(h);
    }
    grid_barrier(bar);
    // ---- P7: outputs from [h_gen; ctx] (overlaps next P1; disjoint buffers) ----
    {
      int idx = wg * 256 + tid;
      if (idx < B_ * NMEL_){
        int b = idx / NMEL_, m = idx % NMEL_;
        const float* hg = h_gen + (long)b * DEC_;
        const float* cx = ctx_f + (long)b * ENC_;
        float s = bf_[m];
        for (int k = 0; k < DEC_; ++k) s += hg[k] * Wf[(long)k * NMEL_ + m];
        for (int k = 0; k < ENC_; ++k) s += cx[k] * Wf[(long)(DEC_ + k) * NMEL_ + m];
        out_spec[((long)b * T_ + t) * NMEL_ + m] = s;
      } else if (idx < B_ * NMEL_ + B_){
        int b = idx - B_ * NMEL_;
        const float* hg = h_gen + (long)b * DEC_;
        const float* cx = ctx_f + (long)b * ENC_;
        float s = bs_[0];
        for (int k = 0; k < DEC_; ++k) s += hg[k] * Ws[k];
        for (int k = 0; k < ENC_; ++k) s += cx[k] * Ws[DEC_ + k];
        out_stop[(long)b * T_ + t] = s;
      }
    }
  }
}

extern "C" void kernel_launch(void* const* d_in, const int* in_sizes, int n_in,
                              void* d_out, int out_size, void* d_ws, size_t ws_size,
                              hipStream_t stream){
  (void)in_sizes; (void)n_in; (void)out_size; (void)ws_size;
  const float* enc      = (const float*)d_in[0];
  const int*   lens     = (const int*)d_in[1];
  const float* target   = (const float*)d_in[2];
  const float* Wp1      = (const float*)d_in[3];
  const float* bp1      = (const float*)d_in[4];
  const float* Wp2      = (const float*)d_in[5];
  const float* bp2      = (const float*)d_in[6];
  const float* Wx_att   = (const float*)d_in[7];
  const float* Wh_att   = (const float*)d_in[8];
  const float* b_att    = (const float*)d_in[9];
  const float* Wx_gen   = (const float*)d_in[10];
  const float* Wh_gen   = (const float*)d_in[11];
  const float* b_gen    = (const float*)d_in[12];
  const float* Wq       = (const float*)d_in[13];
  const float* Wm       = (const float*)d_in[14];
  const float* loc_conv = (const float*)d_in[15];
  const float* Wl       = (const float*)d_in[16];
  const float* v_att    = (const float*)d_in[17];
  const float* Wf       = (const float*)d_in[18];
  const float* bf_      = (const float*)d_in[19];
  const float* Ws       = (const float*)d_in[20];
  const float* bs_      = (const float*)d_in[21];

  char* ws = (char*)d_ws;
  size_t off = 0;
  auto alloc = [&](size_t bytes)->char*{
    char* p = ws + off;
    off = (off + bytes + 255) & ~(size_t)255;
    return p;
  };
  unsigned short* pWx_att = (unsigned short*)alloc(24l * 256 * 512 * 2);
  unsigned short* pWh_att = (unsigned short*)alloc(32l * 256 * 512 * 2);
  unsigned short* pWx_gen = (unsigned short*)alloc(48l * 256 * 512 * 2);
  unsigned short* pWh_gen = (unsigned short*)alloc(32l * 256 * 512 * 2);
  unsigned short* pWp1    = (unsigned short*)alloc(3l * 16 * 512 * 2);
  unsigned short* pWp2    = (unsigned short*)alloc(8l * 16 * 512 * 2);
  unsigned short* pWm     = (unsigned short*)alloc(16l * 8 * 512 * 2);
  unsigned short* tpre    = (unsigned short*)alloc((long)T_ * B_ * PRE_ * 2);
  float* pmem_t           = (float*)alloc((long)B_ * ATT_ * L_ * 4);
  float* gates            = (float*)alloc((long)B_ * 4096 * 4);
  // carry region (zeroed every launch; includes barrier words)
  char* carry = ws + off;
  unsigned* bar           = (unsigned*)alloc(256);
  float* h_att            = (float*)alloc((long)B_ * DEC_ * 4);
  float* c_att            = (float*)alloc((long)B_ * DEC_ * 4);
  float* h_gen            = (float*)alloc((long)B_ * DEC_ * 4);
  float* c_gen            = (float*)alloc((long)B_ * DEC_ * 4);
  unsigned short* hatt_bf = (unsigned short*)alloc((long)B_ * DEC_ * 2);
  unsigned short* hgen_bf = (unsigned short*)alloc((long)B_ * DEC_ * 2);
  float* ctx_f            = (float*)alloc((long)B_ * ENC_ * 4);
  unsigned short* ctx_bf  = (unsigned short*)alloc((long)B_ * ENC_ * 2);
  float* prev_w           = (float*)alloc((long)B_ * L_ * 4);
  float* cum_w            = (float*)alloc((long)B_ * L_ * 4);
  long carry_words = (long)((ws + off) - carry) / 4;

  float* out_spec  = (float*)d_out;
  float* out_stop  = out_spec + (long)B_ * T_ * NMEL_;
  float* out_align = out_stop + (long)B_ * T_;

  auto cdiv = [](long a, long b)->int{ return (int)((a + b - 1) / b); };

  pack_w_bf16<<<cdiv(24l*256*512, 256), 256, 0, stream>>>(Wx_att, pWx_att, 768, 768, 4096);
  pack_w_bf16<<<cdiv(32l*256*512, 256), 256, 0, stream>>>(Wh_att, pWh_att, 1024, 1024, 4096);
  pack_w_bf16<<<cdiv(48l*256*512, 256), 256, 0, stream>>>(Wx_gen, pWx_gen, 1536, 1536, 4096);
  pack_w_bf16<<<cdiv(32l*256*512, 256), 256, 0, stream>>>(Wh_gen, pWh_gen, 1024, 1024, 4096);
  pack_w_bf16<<<cdiv(3l*16*512, 256), 256, 0, stream>>>(Wp1, pWp1, 96, 80, 256);
  pack_w_bf16<<<cdiv(8l*16*512, 256), 256, 0, stream>>>(Wp2, pWp2, 256, 256, 256);
  pack_w_bf16<<<cdiv(16l*8*512, 256), 256, 0, stream>>>(Wm, pWm, 512, 512, 128);
  zero_kernel<<<cdiv(carry_words, 256), 256, 0, stream>>>((unsigned*)carry, carry_words);
  prenet_kernel<<<(B_ * T_) / 16, 256, 0, stream>>>(target, bp1, bp2, pWp1, pWp2, tpre);
  pmem_kernel<<<(B_ * L_) / 16, 256, 0, stream>>>(enc, pWm, pmem_t);
  decoder_kernel<<<GRID_, BLOCK_, 0, stream>>>(
      enc, lens, b_att, b_gen, Wq, loc_conv, Wl, v_att, Wf, bf_, Ws, bs_,
      pWx_att, pWh_att, pWx_gen, pWh_gen, tpre, pmem_t,
      gates, h_att, c_att, h_gen, c_gen, hatt_bf, hgen_bf,
      ctx_f, ctx_bf, prev_w, cum_w, bar,
      out_spec, out_stop, out_align);
}
